// ARConv_Block_8899172237964
// MI455X (gfx1250) — matrix operand order repaired and checked by reading, not compiled, not run on hardware
//
#include <hip/hip_runtime.h>
#include <hip/hip_bf16.h>

// ---------------------------------------------------------------------------
// ARConv block for MI455X (gfx1250, wave32, WMMA).
// Dominant GEMM: einsum 'bckhw,ock->bohw' == M=65536, N=64, K=576 per layer,
// executed with v_wmma_f32_16x16x32_f16 (f16 in, f32 accumulate), double-
// buffered fragments. Gather stage uses paired float2 corner loads with
// border clamp folded into precomputed weights.
// ---------------------------------------------------------------------------

typedef __attribute__((ext_vector_type(16))) _Float16 v16h;
typedef __attribute__((ext_vector_type(8)))  _Float16 v8h;
typedef __attribute__((ext_vector_type(8)))  float    v8f;

#define BB 4
#define CC 64
#define HH 128
#define WW 128
#define HW (HH * WW)          // 16384
#define KK 576                // C * 9
#define TM 32                 // pixels per workgroup tile (main kernel)
#define APAD 8                // pad A rows to 584 halves (1168 B, 16B multiple)
#define AROW (KK + APAD)

// ---------------------------------------------------------------------------
// Repack conv weights W[o][c][3][3] (f32) -> Wh[o][q*64 + c] (f16), row-major
// with 576-half rows so B-fragments are two 16-byte lane loads.
// ---------------------------------------------------------------------------
__global__ __launch_bounds__(256) void prep_weights_kernel(
    const float* __restrict__ W1, const float* __restrict__ W2,
    _Float16* __restrict__ Wh1, _Float16* __restrict__ Wh2) {
  int j = blockIdx.x * 256 + threadIdx.x;
  const int per = CC * KK;            // 36864
  if (j >= 2 * per) return;
  int which = j / per;
  int r = j % per;
  int o = r / KK;
  int kk = r % KK;
  int q = kk / CC;                    // tap index (ky*3+kx)
  int c = kk % CC;
  const float* W = which ? W2 : W1;
  _Float16* Wh = which ? Wh2 : Wh1;
  Wh[o * KK + kk] = (_Float16)W[(o * CC + c) * 9 + q];
}

// ---------------------------------------------------------------------------
// Offset conv (tiled): 3x3 SAME conv (64 -> 2ch) + bias + sigmoid -> rf_h/rf_w.
// 16x16 pixel tile per block; 18x18 halo patch staged in LDS per channel,
// offset weights staged in LDS once. ~7x fewer global loads than direct form.
// ---------------------------------------------------------------------------
#define OT 16                 // offset-conv tile edge
#define PATCH (OT + 2)        // 18
#define PSTR  (PATCH + 2)     // 20 floats per patch row (bank spread)

__global__ __launch_bounds__(256) void offset_conv_kernel(
    const float* __restrict__ in, const float* __restrict__ wof,
    const float* __restrict__ bof, const int* __restrict__ lo_p,
    const int* __restrict__ hi_p, float* __restrict__ rfh,
    float* __restrict__ rfw) {
  __shared__ float patch[PATCH * PSTR];
  __shared__ float wl[2 * CC * 9];    // 4.6 KB offset weights

  const int t = threadIdx.x;
  const int b  = blockIdx.z;
  const int h0 = blockIdx.y * OT;
  const int w0 = blockIdx.x * OT;
  const int ly = t >> 4;
  const int lx = t & 15;

  // stage offset weights once
  for (int e = t; e < 2 * CC * 9; e += 256) wl[e] = wof[e];

  float a0 = bof[0];
  float a1 = bof[1];

  for (int c = 0; c < CC; ++c) {
    const float* plane = in + ((size_t)(b * CC + c)) * HW;
    __syncthreads();                  // previous compute done with patch
    for (int e = t; e < PATCH * PATCH; e += 256) {
      int py = e / PATCH, px = e % PATCH;
      int gy = h0 - 1 + py;
      int gx = w0 - 1 + px;
      float v = 0.0f;
      if (gy >= 0 && gy < HH && gx >= 0 && gx < WW) v = plane[gy * WW + gx];
      patch[py * PSTR + px] = v;
    }
    __syncthreads();
    const float* wc0 = &wl[c * 9];
    const float* wc1 = &wl[(CC + c) * 9];
#pragma unroll
    for (int ky = 0; ky < 3; ++ky) {
#pragma unroll
      for (int kx = 0; kx < 3; ++kx) {
        float v = patch[(ly + ky) * PSTR + (lx + kx)];
        a0 += v * wc0[ky * 3 + kx];
        a1 += v * wc1[ky * 3 + kx];
      }
    }
  }

  float lo = (float)(*lo_p);
  float hi = (float)(*hi_p);
  float s0 = 1.0f / (1.0f + __expf(-a0));
  float s1 = 1.0f / (1.0f + __expf(-a1));
  int p = b * HW + (h0 + ly) * WW + (w0 + lx);
  rfh[p] = lo + (hi - lo) * s0;
  rfw[p] = lo + (hi - lo) * s1;
}

// ---------------------------------------------------------------------------
// Main ARConv kernel: per workgroup a TM=32-pixel row segment x all 64 outputs.
//  stage 1: 32x9 bilinear coords -> LDS; x-border clamp folded into weights
//           so each tap is two float2 loads + 4 FMAs, branch-free
//  stage 2: gather + blend -> f16 A panel in LDS, kk = q*64 + c
//  stage 3: 8 waves, each one 16x16 tile, 18x v_wmma_f32_16x16x32_f16 with
//           double-buffered fragments (loads overlap WMMA)
//  epilogue: + bias, ReLU (mode 0) or + residual (mode 1)
// ---------------------------------------------------------------------------
__global__ __launch_bounds__(256) void arconv_main_kernel(
    const float* __restrict__ in, const float* __restrict__ x_res,
    const float* __restrict__ rfh, const float* __restrict__ rfw,
    const _Float16* __restrict__ Wh, const float* __restrict__ bias,
    float* __restrict__ out, int mode) {
  __shared__ __align__(16) _Float16 A_lds[TM][AROW];  // 37.4 KB
  __shared__ int2   c_off2[TM * 9];                   // top/bot pair offsets
  __shared__ float4 c_wt4[TM * 9];                    // adjusted corner weights

  const int t = threadIdx.x;
  const int pbase = blockIdx.x * TM;
  const int b = pbase >> 14;              // / HW
  const int rem = pbase & (HW - 1);
  const int hh = rem >> 7;                // / W
  const int w0 = rem & (WW - 1);          // 32-aligned, row-contained

  // ---- stage 1: bilinear coordinates (depend only on pixel & tap) ----
  for (int e = t; e < TM * 9; e += 256) {
    int px = e / 9;
    int q = e % 9;
    int p = b * HW + hh * WW + (w0 + px);
    float rh = rfh[p] * 0.5f;
    float rw = rfw[p] * 0.5f;
    float dy = (float)(q / 3 - 1);
    float dx = (float)(q % 3 - 1);
    float ys = (float)hh + dy * rh;
    float xs = (float)(w0 + px) + dx * rw;
    float fy = floorf(ys), fx = floorf(xs);
    float ty = ys - fy, tx = xs - fx;
    int y0 = min(max((int)fy, 0), HH - 1);
    int y1 = min(y0 + 1, HH - 1);
    int x0 = min(max((int)fx, 0), WW - 1);
    float w00 = (1.0f - ty) * (1.0f - tx);
    float w01 = (1.0f - ty) * tx;
    float w10 = ty * (1.0f - tx);
    float w11 = ty * tx;
    // Fold x-border clamp (x1==x0==127) into weights on a pair load at xp.
    bool xe = (x0 >= WW - 1);
    int xp = xe ? (WW - 2) : x0;
    float4 wt;
    wt.x = xe ? 0.0f : w00;
    wt.y = xe ? (w00 + w01) : w01;
    wt.z = xe ? 0.0f : w10;
    wt.w = xe ? (w10 + w11) : w11;
    c_wt4[e] = wt;
    c_off2[e] = make_int2(y0 * WW + xp, y1 * WW + xp);
  }
  __syncthreads();

  // ---- stage 2: paired gather + blend into f16 A panel ----
  {
    int c = t & 63;
    int pxg = t >> 6;                     // 0..3
    const float* plane = in + ((size_t)(b * CC + c)) * HW;
#pragma unroll
    for (int i = 0; i < TM / 4; ++i) {    // 8 pixels per thread
      int px = pxg * (TM / 4) + i;
#pragma unroll
      for (int q = 0; q < 9; ++q) {
        int e = px * 9 + q;
        int2 off = c_off2[e];             // one ds_load_b64
        float4 wt = c_wt4[e];             // one ds_load_b128
        float2 top = *(const float2*)(plane + off.x);  // global_load_b64
        float2 bot = *(const float2*)(plane + off.y);  // global_load_b64
        float v = wt.x * top.x + wt.y * top.y + wt.z * bot.x + wt.w * bot.y;
        A_lds[px][q * CC + c] = (_Float16)v;
      }
    }
  }
  __syncthreads();

  // ---- stage 3: WMMA GEMM. 8 waves -> 2 M-tiles x 4 N-tiles ----
  const int wave = t >> 5;
  const int lane = t & 31;
  const int mt = wave & 1;                // M-tile
  const int nt = wave >> 1;               // N-tile
  const int lhalf = lane >> 4;
  const int l15 = lane & 15;

  v8f acc = {};
  // A fragment (16-bit A 16x32 layout): lane group 0 holds K {0-7,16-23},
  // group 1 holds K {8-15,24-31}; each group = two contiguous b128 LDS loads.
  const _Float16* aptr = &A_lds[mt * 16 + l15][lhalf * 8];
  // B fragment: lane = N column, same K order as A ({0-7,16-23}/{8-15,24-31}) -> two b128 loads.
  const _Float16* bptr = Wh + (size_t)(nt * 16 + l15) * KK + lhalf * 8;

  // Double-buffered fragments: next step's loads issue before current WMMA.
  v8h alo[2], ahi[2], blo[2], bhi[2];
  alo[0] = *(const v8h*)(aptr);
  ahi[0] = *(const v8h*)(aptr + 16);
  blo[0] = *(const v8h*)(bptr);
  bhi[0] = *(const v8h*)(bptr + 16);

#pragma unroll
  for (int kb = 0; kb < KK / 32; ++kb) {  // 18 WMMA per wave-tile
    const int cur = kb & 1;
    const int nxt = cur ^ 1;
    if (kb + 1 < KK / 32) {
      int k0 = (kb + 1) * 32;
      alo[nxt] = *(const v8h*)(aptr + k0);
      ahi[nxt] = *(const v8h*)(aptr + k0 + 16);
      blo[nxt] = *(const v8h*)(bptr + k0);
      bhi[nxt] = *(const v8h*)(bptr + k0 + 16);
    }
    v16h a = __builtin_shufflevector(alo[cur], ahi[cur], 0, 1, 2, 3, 4, 5, 6, 7,
                                     8, 9, 10, 11, 12, 13, 14, 15);
    v16h bm = __builtin_shufflevector(blo[cur], bhi[cur], 0, 1, 2, 3, 4, 5, 6, 7,
                                      8, 9, 10, 11, 12, 13, 14, 15);
    acc = __builtin_amdgcn_wmma_f32_16x16x32_f16(
        false, a, false, bm, (short)0, acc, false, false);
  }

  // ---- epilogue: C/D layout N = lane&15, M = vgpr + 8*(lane>>4) ----
  const int o = nt * 16 + l15;
  const float bval = bias[o];
  const size_t obase = ((size_t)(b * CC + o)) * HW + hh * WW + w0;
#pragma unroll
  for (int v = 0; v < 8; ++v) {
    int px = mt * 16 + v + 8 * lhalf;
    float val = acc[v] + bval;
    if (mode == 0) {
      val = fmaxf(val, 0.0f);             // layer-1 ReLU
    } else {
      val += x_res[obase + px];           // layer-2 residual
    }
    out[obase + px] = val;
  }
}

// ---------------------------------------------------------------------------
// Host-side launch (graph-capture safe: only async launches on `stream`).
// ---------------------------------------------------------------------------
extern "C" void kernel_launch(void* const* d_in, const int* in_sizes, int n_in,
                              void* d_out, int out_size, void* d_ws, size_t ws_size,
                              hipStream_t stream) {
  (void)in_sizes; (void)n_in; (void)out_size; (void)ws_size;

  const float* x      = (const float*)d_in[0];
  const float* w_off1 = (const float*)d_in[1];
  const float* b_off1 = (const float*)d_in[2];
  const float* W1     = (const float*)d_in[3];
  const float* b1     = (const float*)d_in[4];
  const float* w_off2 = (const float*)d_in[5];
  const float* b_off2 = (const float*)d_in[6];
  const float* W2     = (const float*)d_in[7];
  const float* b2     = (const float*)d_in[8];
  // d_in[9] = epochs (unused by reference math)
  const int* hw_lo    = (const int*)d_in[10];
  const int* hw_hi    = (const int*)d_in[11];
  float* out          = (float*)d_out;

  // Workspace layout
  const size_t nAct = (size_t)BB * CC * HW;       // 4M floats
  const size_t nPix = (size_t)BB * HW;            // 64K floats
  float* y1  = (float*)d_ws;
  float* rfh = y1 + nAct;
  float* rfw = rfh + nPix;
  _Float16* Wh1 = (_Float16*)(rfw + nPix);
  _Float16* Wh2 = Wh1 + (size_t)CC * KK;

  const int nPixTotal = BB * HW;                  // 65536
  const int gemmBlocks = nPixTotal / TM;          // 2048
  dim3 offGrid(WW / OT, HH / OT, BB);             // 8 x 8 x 4

  // Weight repack (both layers)
  prep_weights_kernel<<<(2 * CC * KK + 255) / 256, 256, 0, stream>>>(
      W1, W2, Wh1, Wh2);

  // Layer 1
  offset_conv_kernel<<<offGrid, 256, 0, stream>>>(
      x, w_off1, b_off1, hw_lo, hw_hi, rfh, rfw);
  arconv_main_kernel<<<gemmBlocks, 256, 0, stream>>>(
      x, x, rfh, rfw, Wh1, b1, y1, /*mode=*/0);

  // Layer 2 (+ residual with original x)
  offset_conv_kernel<<<offGrid, 256, 0, stream>>>(
      y1, w_off2, b_off2, hw_lo, hw_hi, rfh, rfw);
  arconv_main_kernel<<<gemmBlocks, 256, 0, stream>>>(
      y1, x, rfh, rfw, Wh2, b2, out, /*mode=*/1);
}
